// QuantizedLinear_4964982194264
// MI455X (gfx1250) — compile-verified, matrix operand order checked by reading, not run on hardware
//
#include <hip/hip_runtime.h>

// ---------------------------------------------------------------------------
// W4A16 quantized GEMM for MI455X (gfx1250, wave32, WMMA).
//   y[m,n] = sum_k x[m,k] * (q[n,k] * scale[n, k/128]) + bias[n]
// M = 8192, K = 4096, N = 11008.
// Compute-bound (0.74 TFLOP vs ~0.7 GB min traffic). Strategy:
//   pass 1: x fp32 -> f16 (once),  qweight int4*scale -> f16 (once) into d_ws
//   pass 2: pure f16 WMMA GEMM, f32 accumulate. Inner loop = WMMA + LDS only.
// Fallback (d_ws too small): fused-dequant f16 WMMA GEMM (round-1 structure).
// ---------------------------------------------------------------------------

typedef __attribute__((ext_vector_type(16))) _Float16 v16h;
typedef __attribute__((ext_vector_type(8)))  _Float16 v8h;
typedef __attribute__((ext_vector_type(4)))  _Float16 v4h;
typedef __attribute__((ext_vector_type(8)))  float    v8f;
typedef __attribute__((ext_vector_type(4)))  float    v4f;
typedef __attribute__((ext_vector_type(4)))  int      v4i;

#define BKS   64          // K per LDS stage (= 2 WMMA k-steps of 32)
#define PADH  8           // f16 pad -> 144B row stride, conflict-free frag loads
#define LDKH  (BKS + PADH)

// ---------------------------------------------------------------------------
// Pass-1 kernels
// ---------------------------------------------------------------------------
__global__ __launch_bounds__(256)
void convert_x_f16(const float* __restrict__ X, _Float16* __restrict__ Xh,
                   long long total) {
  long long i = ((long long)blockIdx.x * blockDim.x + threadIdx.x) * 8;
  if (i >= total) return;
  v4f a = *(const v4f*)(X + i);
  v4f b = *(const v4f*)(X + i + 4);
  v8h o;
  o[0] = (_Float16)a.x; o[1] = (_Float16)a.y; o[2] = (_Float16)a.z; o[3] = (_Float16)a.w;
  o[4] = (_Float16)b.x; o[5] = (_Float16)b.y; o[6] = (_Float16)b.z; o[7] = (_Float16)b.w;
  *(v8h*)(Xh + i) = o;
}

__global__ __launch_bounds__(256)
void dequant_w_f16(const int* __restrict__ Q, const float* __restrict__ S,
                   _Float16* __restrict__ Wh, int K, int KG, long long total) {
  long long i = ((long long)blockIdx.x * blockDim.x + threadIdx.x) * 8;
  if (i >= total) return;
  const long long n = i / K;
  const int k = (int)(i - n * K);           // 8-aligned, 8 | 128 -> one group
  const float s = S[n * KG + (k >> 7)];
  v4i a = *(const v4i*)(Q + i);
  v4i b = *(const v4i*)(Q + i + 4);
  v8h o;
  o[0] = (_Float16)((float)a.x * s); o[1] = (_Float16)((float)a.y * s);
  o[2] = (_Float16)((float)a.z * s); o[3] = (_Float16)((float)a.w * s);
  o[4] = (_Float16)((float)b.x * s); o[5] = (_Float16)((float)b.y * s);
  o[6] = (_Float16)((float)b.z * s); o[7] = (_Float16)((float)b.w * s);
  *(v8h*)(Wh + i) = o;
}

// ---------------------------------------------------------------------------
// Helper: assemble a 16x32 f16 WMMA fragment from LDS (ISA 16-bit layout:
// lane<16 -> K base 0, lane>=16 -> K base 8; elems 0..7 = Kb..Kb+7,
// elems 8..15 = Kb+16..Kb+23).
// ---------------------------------------------------------------------------
__device__ __forceinline__ v16h load_frag(const _Float16* rowPtr, int kb) {
  v8h lo = *(const v8h*)(rowPtr + kb);
  v8h hi = *(const v8h*)(rowPtr + kb + 16);
  return __builtin_shufflevector(lo, hi, 0, 1, 2, 3, 4, 5, 6, 7,
                                         8, 9, 10, 11, 12, 13, 14, 15);
}

// ---------------------------------------------------------------------------
// Pass-2: pure f16 GEMM.  Block tile 256(M) x 128(N), 8 waves (wave32),
// wave grid 4(M) x 2(N), wave tile 64x64 = 4x4 WMMA accumulators.
// Per stage (K=64): 32 ds_load_b128 + 32 v_wmma per wave.
// ---------------------------------------------------------------------------
#define GBM 256
#define GBN 128

__global__ __launch_bounds__(256)
void f16_wmma_gemm(const _Float16* __restrict__ Xh,  // [M, K]
                   const _Float16* __restrict__ Wh,  // [N, K]
                   const float*    __restrict__ Bv,  // [N]
                   float*          __restrict__ Y,   // [M, N]
                   int M, int K, int N) {
  __shared__ _Float16 lA[GBM][LDKH];   // 256*72*2 = 36 KB
  __shared__ _Float16 lB[GBN][LDKH];   // 128*72*2 = 18 KB

  const int tid  = threadIdx.x;
  const int lane = tid & 31;
  const int wave = tid >> 5;
  const int wm   = wave & 3;    // 4 waves along M
  const int wn   = wave >> 2;   // 2 waves along N

  const int bm = blockIdx.y * GBM;
  const int bn = blockIdx.x * GBN;

  // staging: 16-byte chunks (8 f16). A: 2048 chunks -> 8/thread; B: 4/thread.
  const int arow = tid >> 3;          // 0..31 (A rows advance by 32 per chunk l)
  const int acol = (tid & 7) * 8;     // 0,8,...,56

  const int nStages = K / BKS;

  v8f acc[4][4];
#pragma unroll
  for (int i = 0; i < 4; ++i)
#pragma unroll
    for (int j = 0; j < 4; ++j)
      acc[i][j] = (v8f)0.0f;

  v8h aStg[8];
  v8h bStg[4];

  // ---- stage 0 global loads ----
#pragma unroll
  for (int l = 0; l < 8; ++l)
    aStg[l] = *(const v8h*)(Xh + (size_t)(bm + arow + l * 32) * K + acol);
#pragma unroll
  for (int l = 0; l < 4; ++l)
    bStg[l] = *(const v8h*)(Wh + (size_t)(bn + arow + l * 32) * K + acol);

  for (int kt = 0; kt < nStages; ++kt) {
    // ---- commit staged registers to LDS (pure copy, ds_store_b128) ----
#pragma unroll
    for (int l = 0; l < 8; ++l)
      *(v8h*)&lA[arow + l * 32][acol] = aStg[l];
#pragma unroll
    for (int l = 0; l < 4; ++l)
      *(v8h*)&lB[arow + l * 32][acol] = bStg[l];
    __syncthreads();

    // ---- issue next stage's global loads before compute ----
    if (kt + 1 < nStages) {
      const int k0n = (kt + 1) * BKS;
#pragma unroll
      for (int l = 0; l < 8; ++l)
        aStg[l] = *(const v8h*)(Xh + (size_t)(bm + arow + l * 32) * K + k0n + acol);
#pragma unroll
      for (int l = 0; l < 4; ++l)
        bStg[l] = *(const v8h*)(Wh + (size_t)(bn + arow + l * 32) * K + k0n + acol);
    }
    // ---- L2 prefetch of stage kt+2 (global_prefetch_b8) ----
    if (kt + 2 < nStages) {
      const int k0p = (kt + 2) * BKS;
      __builtin_prefetch(Xh + (size_t)(bm + arow) * K + k0p + acol, 0, 1);
      __builtin_prefetch(Wh + (size_t)(bn + arow) * K + k0p + acol, 0, 1);
    }

    // ---- compute: 2 WMMA k-substeps of 32 ----
#pragma unroll
    for (int ks = 0; ks < 2; ++ks) {
      const int kb = ks * 32 + ((lane & 16) ? 8 : 0);
      const int lr = lane & 15;

      v16h af[4];
#pragma unroll
      for (int mi = 0; mi < 4; ++mi)
        af[mi] = load_frag(&lA[wm * 64 + mi * 16 + lr][0], kb);

#pragma unroll
      for (int ni = 0; ni < 4; ++ni) {
        v16h bf = load_frag(&lB[wn * 64 + ni * 16 + lr][0], kb);
#pragma unroll
        for (int mi = 0; mi < 4; ++mi) {
          acc[mi][ni] = __builtin_amdgcn_wmma_f32_16x16x32_f16(
              false, af[mi], false, bf, (short)0, acc[mi][ni], false, false);
        }
      }
    }
    __syncthreads();
  }

  // ---- epilogue: bias + store (C layout: VGPR r -> M=r+(lane>=16?8:0), N=lane%16)
  const int moff = (lane & 16) ? 8 : 0;
  const int lr   = lane & 15;
#pragma unroll
  for (int ni = 0; ni < 4; ++ni) {
    const int col = bn + wn * 64 + ni * 16 + lr;
    const float bias = Bv[col];
#pragma unroll
    for (int mi = 0; mi < 4; ++mi) {
      const int mbase = bm + wm * 64 + mi * 16 + moff;
#pragma unroll
      for (int r = 0; r < 8; ++r)
        Y[(size_t)(mbase + r) * N + col] = acc[mi][ni][r] + bias;
    }
  }
}

// ---------------------------------------------------------------------------
// Fallback: fused dequant + f16 WMMA GEMM (block 128x128, wave tile 64x32)
// Used only if d_ws cannot hold the f16 copies.
// ---------------------------------------------------------------------------
#define FBM 128
#define FBN 128

__global__ __launch_bounds__(256)
void w4a16_fused_gemm(const float* __restrict__ X, const int* __restrict__ Q,
                      const float* __restrict__ S, const float* __restrict__ Bv,
                      float* __restrict__ Y, int M, int K, int N) {
  __shared__ _Float16 lA[FBM][LDKH];
  __shared__ _Float16 lB[FBN][LDKH];

  const int tid  = threadIdx.x;
  const int lane = tid & 31;
  const int wave = tid >> 5;
  const int wm   = wave & 1;
  const int wn   = wave >> 1;
  const int bm = blockIdx.y * FBM;
  const int bn = blockIdx.x * FBN;
  const int lrow = tid >> 4;
  const int lcol = (tid & 15) * 4;
  const int KG = K >> 7;
  const int nStages = K / BKS;

  v8f acc[4][2];
#pragma unroll
  for (int i = 0; i < 4; ++i)
#pragma unroll
    for (int j = 0; j < 2; ++j) acc[i][j] = (v8f)0.0f;

  v4f aReg[8]; v4i bReg[8]; float sReg[8];
#pragma unroll
  for (int l = 0; l < 8; ++l) {
    const int r = lrow + l * 16;
    aReg[l] = *(const v4f*)(X + (size_t)(bm + r) * K + lcol);
    bReg[l] = *(const v4i*)(Q + (size_t)(bn + r) * K + lcol);
    sReg[l] = S[(size_t)(bn + r) * KG];
  }

  for (int kt = 0; kt < nStages; ++kt) {
#pragma unroll
    for (int l = 0; l < 8; ++l) {
      const int r = lrow + l * 16;
      v4h ap;
      ap[0] = (_Float16)aReg[l].x; ap[1] = (_Float16)aReg[l].y;
      ap[2] = (_Float16)aReg[l].z; ap[3] = (_Float16)aReg[l].w;
      *(v4h*)&lA[r][lcol] = ap;
      const float s = sReg[l];
      v4h bp;
      bp[0] = (_Float16)((float)bReg[l].x * s);
      bp[1] = (_Float16)((float)bReg[l].y * s);
      bp[2] = (_Float16)((float)bReg[l].z * s);
      bp[3] = (_Float16)((float)bReg[l].w * s);
      *(v4h*)&lB[r][lcol] = bp;
    }
    __syncthreads();

    if (kt + 1 < nStages) {
      const int k0n = (kt + 1) * BKS;
#pragma unroll
      for (int l = 0; l < 8; ++l) {
        const int r = lrow + l * 16;
        aReg[l] = *(const v4f*)(X + (size_t)(bm + r) * K + k0n + lcol);
        bReg[l] = *(const v4i*)(Q + (size_t)(bn + r) * K + k0n + lcol);
        sReg[l] = S[(size_t)(bn + r) * KG + (k0n >> 7)];
      }
    }

#pragma unroll
    for (int ks = 0; ks < 2; ++ks) {
      const int kb = ks * 32 + ((lane & 16) ? 8 : 0);
      const int lr = lane & 15;
      v16h af[4];
#pragma unroll
      for (int mi = 0; mi < 4; ++mi)
        af[mi] = load_frag(&lA[wm * 64 + mi * 16 + lr][0], kb);
#pragma unroll
      for (int ni = 0; ni < 2; ++ni) {
        v16h bf = load_frag(&lB[wn * 32 + ni * 16 + lr][0], kb);
#pragma unroll
        for (int mi = 0; mi < 4; ++mi)
          acc[mi][ni] = __builtin_amdgcn_wmma_f32_16x16x32_f16(
              false, af[mi], false, bf, (short)0, acc[mi][ni], false, false);
      }
    }
    __syncthreads();
  }

  const int moff = (lane & 16) ? 8 : 0;
  const int lr   = lane & 15;
#pragma unroll
  for (int ni = 0; ni < 2; ++ni) {
    const int col = bn + wn * 32 + ni * 16 + lr;
    const float bias = Bv[col];
#pragma unroll
    for (int mi = 0; mi < 4; ++mi) {
      const int mbase = bm + wm * 64 + mi * 16 + moff;
#pragma unroll
      for (int r = 0; r < 8; ++r)
        Y[(size_t)(mbase + r) * N + col] = acc[mi][ni][r] + bias;
    }
  }
}

// ---------------------------------------------------------------------------
extern "C" void kernel_launch(void* const* d_in, const int* in_sizes, int n_in,
                              void* d_out, int out_size, void* d_ws, size_t ws_size,
                              hipStream_t stream) {
  const float* x      = (const float*)d_in[0];
  const int*   qw     = (const int*)  d_in[1];
  const float* scales = (const float*)d_in[2];
  const float* bias   = (const float*)d_in[3];
  float*       out    = (float*)d_out;

  const int O = in_sizes[3];                 // 11008
  const int I = in_sizes[1] / O;             // 4096
  const int M = in_sizes[0] / I;             // 8192
  const int KG = I >> 7;

  const size_t xBytes = (size_t)M * I * sizeof(_Float16);
  const size_t wOff   = (xBytes + 255) & ~(size_t)255;
  const size_t wBytes = (size_t)O * I * sizeof(_Float16);
  const size_t need   = wOff + wBytes;

  if (ws_size >= need && (M % GBM) == 0 && (O % GBN) == 0 && (I % BKS) == 0) {
    _Float16* xh = (_Float16*)d_ws;
    _Float16* wh = (_Float16*)((char*)d_ws + wOff);

    const long long xTot = (long long)M * I;
    const long long wTot = (long long)O * I;
    convert_x_f16<<<dim3((unsigned)((xTot / 8 + 255) / 256)), dim3(256), 0, stream>>>(
        x, xh, xTot);
    dequant_w_f16<<<dim3((unsigned)((wTot / 8 + 255) / 256)), dim3(256), 0, stream>>>(
        qw, scales, wh, I, KG, wTot);

    dim3 grid(O / GBN, M / GBM);             // 86 x 32
    f16_wmma_gemm<<<grid, dim3(256), 0, stream>>>(xh, wh, bias, out, M, I, O);
  } else {
    dim3 grid(O / FBN, M / FBM);             // 86 x 64
    w4a16_fused_gemm<<<grid, dim3(256), 0, stream>>>(x, qw, scales, bias, out, M, I, O);
  }
}
